// PairwiseInteract_24661702213903
// MI455X (gfx1250) — compile-verified, matrix operand order checked by reading, not run on hardware
//
#include <hip/hip_runtime.h>

typedef __attribute__((ext_vector_type(16))) _Float16 v16h;
typedef __attribute__((ext_vector_type(8)))  float    v8f;

#define NN       1024
#define JSPLIT   8
#define JCHUNK   (NN / JSPLIT)   // 128
#define WAVES    4

__device__ __forceinline__ v8f wmma16(v16h a, v16h b, v8f c) {
  return __builtin_amdgcn_wmma_f32_16x16x32_f16(false, a, false, b, (short)0, c,
                                                false, false);
}

// K index inside one 16x32 f16 A/B fragment for half-slot h (0..15) and lane
// group g (0 = lanes 0-15, 1 = lanes 16-31), per CDNA5 ISA 7.12.2.
__device__ __forceinline__ int kmap(int h, int g) {
  return (h & 7) + ((h >> 3) << 4) + (g << 3);
}

// All GEMMs are computed transposed:  h_next^T = W^T @ h^T
//   A operand = W^T fragments (lane = out-feature row, vector = K in-feature)
//   B operand = h^T fragments (lane = pair,            vector = K in-feature)
//   D output  = h_next^T      (reg  = out-feature,     lane   = pair)
// A lane of D holds pair `ln` and out-features {mt*16 + r + g*8}, which is
// exactly the K-set the next layer's B fragment wants in that lane, so the
// D->B conversion is lane-local (single v_max_num + packed cvt, no LDS).
// Bias trick: K slot 50 (inside the 50->64 pad) carries a constant 1.0 through
// every layer (W^T[50][50] = 1), and W^T[row][50] = bias[row].

__global__ __launch_bounds__(WAVES * 32)
void pairwise_force_kernel(const float* __restrict__ obj0,
                           const float* __restrict__ obj1,
                           const float* __restrict__ prev0,
                           const float* __restrict__ prev1,
                           const float* __restrict__ gW0, const float* __restrict__ gb0,
                           const float* __restrict__ gW1, const float* __restrict__ gb1,
                           const float* __restrict__ gW2, const float* __restrict__ gb2,
                           const float* __restrict__ gW3, const float* __restrict__ gb3,
                           float* __restrict__ forces) {
  const int tid  = threadIdx.x;
  const int wave = tid >> 5;
  const int lane = tid & 31;
  const int g    = lane >> 4;
  const int ln   = lane & 15;

  const int blk   = blockIdx.x;             // 1024 blocks
  const int m     = blk >> 7;               // module 0..7
  const int local = blk & 127;
  const int task  = local * WAVES + wave;   // 0..511
  const int itile = task >> 3;              // 0..63 (16 actee rows each)
  const int jch   = task & 7;               // j chunk 0..7
  const int aIdx  = m >> 1;
  const int cIdx  = m & 1;

  const float* actor = (aIdx == 0) ? obj0 : (aIdx == 1) ? obj1
                       : (aIdx == 2) ? prev0 : prev1;
  const float* actee = (cIdx == 0) ? obj0 : obj1;

  const int   i0 = itile * 16;
  const float ai = actee[i0 + ln];          // this lane's pair (B lane = pair)

  // ---- layer-1 per-lane packed weights in B-fragment K order ---------------
  // h1[K] = relu(actor[j]*w0[K] + (actee[i]*w1[K] + b0[K])),  h1[50] = 1.
  v16h w0v[2], vprev[2];
#pragma unroll
  for (int f = 0; f < 2; ++f) {
#pragma unroll
    for (int h = 0; h < 16; ++h) {
      int  K   = f * 32 + kmap(h, g);
      bool ok  = (K < 50);
      bool one = (K == 50);
      int  kk  = ok ? K : 0;
      float w0 = gW0[m * 100 + kk];        // row 0: actor weight
      float w1 = gW0[m * 100 + 50 + kk];   // row 1: actee weight
      float b0 = gb0[m * 50 + kk];
      w0v[f][h]   = ok ? (_Float16)w0 : (_Float16)0.0f;
      vprev[f][h] = ok ? (_Float16)(ai * w1 + b0)
                       : (one ? (_Float16)1.0f : (_Float16)0.0f);
    }
  }

  // ---- A fragments: transposed, padded weights with bias/ones columns ------
  v16h W1T[4][2], W2T[4][2], W3T[2][2];
#pragma unroll
  for (int mt = 0; mt < 4; ++mt) {
#pragma unroll
    for (int ks = 0; ks < 2; ++ks) {
      v16h f1, f2;
#pragma unroll
      for (int h = 0; h < 16; ++h) {
        int  K   = ks * 32 + kmap(h, g);
        int  row = mt * 16 + ln;           // output feature
        bool wv  = (row < 50) && (K < 50);
        bool bv  = (row < 50) && (K == 50);
        bool idv = (row == 50) && (K == 50);
        int  wi  = wv ? (m * 2500 + K * 50 + row) : 0;
        int  bi  = bv ? (m * 50 + row) : 0;
        float w1 = gW1[wi], w2 = gW2[wi];
        float b1 = gb1[bi], b2 = gb2[bi];
        f1[h] = wv ? (_Float16)w1
                   : (bv ? (_Float16)b1 : (idv ? (_Float16)1.0f : (_Float16)0.0f));
        f2[h] = wv ? (_Float16)w2
                   : (bv ? (_Float16)b2 : (idv ? (_Float16)1.0f : (_Float16)0.0f));
      }
      W1T[mt][ks] = f1;
      W2T[mt][ks] = f2;
    }
  }
#pragma unroll
  for (int mt = 0; mt < 2; ++mt) {
#pragma unroll
    for (int ks = 0; ks < 2; ++ks) {
      v16h f3;
#pragma unroll
      for (int h = 0; h < 16; ++h) {
        int  K   = ks * 32 + kmap(h, g);
        int  row = mt * 16 + ln;           // output 0..31 (20 real)
        bool wv  = (row < 20) && (K < 50);
        bool bv  = (row < 20) && (K == 50);
        int  wi  = wv ? (m * 1000 + K * 20 + row) : 0;
        int  bi  = bv ? (m * 20 + row) : 0;
        float w3 = gW3[wi];
        float b3 = gb3[bi];
        f3[h] = wv ? (_Float16)w3 : (bv ? (_Float16)b3 : (_Float16)0.0f);
      }
      W3T[mt][ks] = f3;
    }
  }

  // persistent layer-4 accumulators: D = out^T (reg = out-feature, lane = pair)
  v8f acc[2];
#pragma unroll
  for (int t = 0; t < 2; ++t)
#pragma unroll
    for (int r = 0; r < 8; ++r) acc[t][r] = 0.0f;

  const int jBase = jch * JCHUNK;
#pragma unroll 1
  for (int j = jBase; j < jBase + JCHUNK; ++j) {
    // ---- layer 1: build h1^T B fragments (packed f16 fma + relu) ----
    _Float16 aj = (_Float16)actor[j];
    v16h ajv;
#pragma unroll
    for (int h = 0; h < 16; ++h) ajv[h] = aj;
    v16h B0, B1;
    {
      v16h t0 = ajv * w0v[0] + vprev[0];
      v16h t1 = ajv * w0v[1] + vprev[1];
#pragma unroll
      for (int h = 0; h < 16; ++h) {
        B0[h] = t0[h] > (_Float16)0.0f ? t0[h] : (_Float16)0.0f;
        B1[h] = t1[h] > (_Float16)0.0f ? t1[h] : (_Float16)0.0f;
      }
    }

    // ---- layer 2: h2^T = W1^T @ h1^T; lane-local relu+cvt into next B ----
    v16h nB0, nB1;
#pragma unroll
    for (int mt = 0; mt < 4; ++mt) {
      v8f cz;
#pragma unroll
      for (int r = 0; r < 8; ++r) cz[r] = 0.0f;
      cz = wmma16(W1T[mt][0], B0, cz);
      cz = wmma16(W1T[mt][1], B1, cz);
#pragma unroll
      for (int r = 0; r < 8; ++r) {
        float v = __builtin_fmaxf(cz[r], 0.0f);   // single v_max_num_f32
        if (mt < 2) nB0[(mt & 1) * 8 + r] = (_Float16)v;
        else        nB1[(mt & 1) * 8 + r] = (_Float16)v;
      }
    }
    B0 = nB0;
    B1 = nB1;

    // ---- layer 3: h3^T = W2^T @ h2^T ----
#pragma unroll
    for (int mt = 0; mt < 4; ++mt) {
      v8f cz;
#pragma unroll
      for (int r = 0; r < 8; ++r) cz[r] = 0.0f;
      cz = wmma16(W2T[mt][0], B0, cz);
      cz = wmma16(W2T[mt][1], B1, cz);
#pragma unroll
      for (int r = 0; r < 8; ++r) {
        float v = __builtin_fmaxf(cz[r], 0.0f);
        if (mt < 2) nB0[(mt & 1) * 8 + r] = (_Float16)v;
        else        nB1[(mt & 1) * 8 + r] = (_Float16)v;
      }
    }
    B0 = nB0;
    B1 = nB1;

    // ---- layer 4: out^T += W3^T @ h3^T (bias folded via ones-slot) ----
    acc[0] = wmma16(W3T[0][0], B0, acc[0]);
    acc[0] = wmma16(W3T[0][1], B1, acc[0]);
    acc[1] = wmma16(W3T[1][0], B0, acc[1]);
    acc[1] = wmma16(W3T[1][1], B1, acc[1]);
  }

  // ---- reduce into global forces[c][i][f]: lane holds pair ln, regs = outs --
  float* dst = &forces[(cIdx * NN + i0 + ln) * 20];
#pragma unroll
  for (int mt = 0; mt < 2; ++mt) {
#pragma unroll
    for (int r = 0; r < 8; ++r) {
      int o = mt * 16 + r + g * 8;
      if (o < 20) atomicAdd(dst + o, acc[mt][r]);
    }
  }
}

__global__ void zero_kernel(float* __restrict__ p, int n) {
  int t = blockIdx.x * blockDim.x + threadIdx.x;
  if (t < n) p[t] = 0.0f;
}

__global__ void apply_force_kernel(const float* __restrict__ forces,
                                   const float* __restrict__ aW0,
                                   const float* __restrict__ ab0,
                                   const float* __restrict__ aW1,
                                   const float* __restrict__ ab1,
                                   float* __restrict__ out) {
  int t = blockIdx.x * blockDim.x + threadIdx.x;
  if (t >= 2 * NN) return;
  int cc = t / NN;
  const float* f = forces + t * 20;   // t = cc*NN + i matches output layout
  float o = ab1[cc];
  for (int oo = 0; oo < 50; ++oo) {
    float s = ab0[cc * 50 + oo];
#pragma unroll
    for (int k = 0; k < 20; ++k) s += f[k] * aW0[cc * 1000 + k * 50 + oo];
    s = __builtin_fmaxf(s, 0.0f);
    o += s * aW1[cc * 50 + oo];       // aW1 is [2,50,1]
  }
  out[t] = o;                          // concat(pred0, pred1)
}

extern "C" void kernel_launch(void* const* d_in, const int* in_sizes, int n_in,
                              void* d_out, int out_size, void* d_ws, size_t ws_size,
                              hipStream_t stream) {
  const float* obj0  = (const float*)d_in[0];
  const float* obj1  = (const float*)d_in[1];
  const float* prev0 = (const float*)d_in[2];
  const float* prev1 = (const float*)d_in[3];
  const float* gW0 = (const float*)d_in[4];
  const float* gb0 = (const float*)d_in[5];
  const float* gW1 = (const float*)d_in[6];
  const float* gb1 = (const float*)d_in[7];
  const float* gW2 = (const float*)d_in[8];
  const float* gb2 = (const float*)d_in[9];
  const float* gW3 = (const float*)d_in[10];
  const float* gb3 = (const float*)d_in[11];
  const float* aW0 = (const float*)d_in[12];
  const float* ab0 = (const float*)d_in[13];
  const float* aW1 = (const float*)d_in[14];
  const float* ab1 = (const float*)d_in[15];

  float* forces = (float*)d_ws;             // 2*1024*20 floats = 160 KB
  const int nForces = 2 * NN * 20;

  zero_kernel<<<(nForces + 255) / 256, 256, 0, stream>>>(forces, nForces);

  // 8 modules * 128 blocks; each block = 4 waves, each wave = (itile, jchunk)
  pairwise_force_kernel<<<1024, WAVES * 32, 0, stream>>>(
      obj0, obj1, prev0, prev1,
      gW0, gb0, gW1, gb1, gW2, gb2, gW3, gb3, forces);

  apply_force_kernel<<<(2 * NN + 255) / 256, 256, 0, stream>>>(
      forces, aW0, ab0, aW1, ab1, (float*)d_out);
}
